// TreeGRUBatch_84061099917530
// MI455X (gfx1250) — matrix operand order repaired and checked by reading, not compiled, not run on hardware
//
#include <hip/hip_runtime.h>
#include <hip/hip_bf16.h>

// ---------------------------------------------------------------------------
// TreeGRUBatch for MI455X (gfx1250).
// * All GEMMs via v_wmma_f32_16x16x32_bf16 (fp32 -> bf16 on load).
// * Weights pre-transposed/converted to bf16 (N x Kpad) once per launch.
// * Workgroup = 8 waves computing a 256x64 output tile; each wave owns two
//   16-row strips (32x64, 8 accumulators) so every LDS B-fragment feeds two
//   WMMAs. The shared 64-col weight panel is staged into LDS by the Tensor
//   Data Mover (double-buffered, s_wait_tensorcnt + barriers). TDM pad_enable
//   gives a 272B column stride so ds_load_b128 B reads are conflict-free.
// * Epilogues fused: bias, min-over-P (pred), has_cond scale (bitmap), ReLU.
// ---------------------------------------------------------------------------

typedef __attribute__((ext_vector_type(16))) __bf16 v16bf;
typedef __attribute__((ext_vector_type(8)))  float  v8f;
typedef unsigned int u32x4 __attribute__((ext_vector_type(4)));
typedef int          i32x4 __attribute__((ext_vector_type(4)));
typedef int          i32x8 __attribute__((ext_vector_type(8)));

#define BSZ   4096
#define NODES 15
#define MN    (BSZ * NODES)          // 61440
#define MP    (BSZ * NODES * 2 * 4)  // 491520

static __device__ __forceinline__ float sigmoidf_(float x) {
  return 1.0f / (1.0f + __expf(-x));
}

// ------------------------- weight transpose + bf16 --------------------------
__global__ void k_wT(const float* __restrict__ src, __bf16* __restrict__ dst,
                     int K, int N, int Kpad) {
  int i = blockIdx.x * blockDim.x + threadIdx.x;
  int total = N * Kpad;
  if (i >= total) return;
  int n = i / Kpad, k = i - n * Kpad;
  float v = (k < K) ? src[(size_t)k * N + n] : 0.0f;
  dst[i] = (__bf16)v;
}

// --------------------------- A fragment loader ------------------------------
// A frag (16x32 bf16): lane = row M, elems j: K = 8*hi + (j&7) + (j>=8 ? 16 : 0)
static __device__ __forceinline__ v16bf load_a_frag(const float* __restrict__ arow,
                                                    int k0, int hi) {
  const float4* p0 = reinterpret_cast<const float4*>(arow + k0 + 8 * hi);
  float4 f0 = p0[0];
  float4 f1 = p0[1];
  const float4* p1 = reinterpret_cast<const float4*>(arow + k0 + 16 + 8 * hi);
  float4 f2 = p1[0];
  float4 f3 = p1[1];
  v16bf a;
  a[0] = (__bf16)f0.x;  a[1] = (__bf16)f0.y;  a[2] = (__bf16)f0.z;  a[3] = (__bf16)f0.w;
  a[4] = (__bf16)f1.x;  a[5] = (__bf16)f1.y;  a[6] = (__bf16)f1.z;  a[7] = (__bf16)f1.w;
  a[8] = (__bf16)f2.x;  a[9] = (__bf16)f2.y;  a[10] = (__bf16)f2.z; a[11] = (__bf16)f2.w;
  a[12] = (__bf16)f3.x; a[13] = (__bf16)f3.y; a[14] = (__bf16)f3.z; a[15] = (__bf16)f3.w;
  return a;
}

static __device__ __forceinline__ v16bf load_a_frag_guard(const float* __restrict__ arow,
                                                          int k0, int hi, int Kreal) {
  v16bf a;
#pragma unroll
  for (int j = 0; j < 8; ++j) {
    int ka = k0 + 8 * hi + j;
    int kb = k0 + 16 + 8 * hi + j;
    a[j]     = (__bf16)(ka < Kreal ? arow[ka] : 0.0f);
    a[8 + j] = (__bf16)(kb < Kreal ? arow[kb] : 0.0f);
  }
  return a;
}

// ---------------------- TDM panel load: global -> LDS -----------------------
// 2D tile: dim0 = chunk (k elements, bf16), dim1 = 64 columns, row stride Kpad.
// pad_enable: one 16B pad per (chunk*2)-byte column -> LDS column stride
// chunk*2+16 bytes (conflict-free ds reads).
static __device__ __forceinline__ void tdm_load_panel(unsigned lds_off,
                                                      const __bf16* gsrc,
                                                      int Kpad, int chunkElems,
                                                      unsigned intvCode) {
  unsigned long long ga = (unsigned long long)(uintptr_t)gsrc;
  u32x4 g0;
  g0[0] = 1u;                                    // count=1, user descriptor
  g0[1] = lds_off;                               // LDS byte address
  g0[2] = (unsigned)(ga & 0xFFFFFFFFu);          // global_addr[31:0]
  g0[3] = (unsigned)(ga >> 32) | (2u << 30);     // global_addr[56:32] | type=2
  i32x8 g1;
  g1[0] = (int)((1u << 16)                       // data_size = 2 bytes
              | (1u << 20)                       // pad_enable
              | (intvCode << 22)                 // pad_interval
              | (3u << 25));                     // pad_amount = 4 DWORDs (16B)
  g1[1] = (int)(((unsigned)chunkElems & 0xFFFFu) << 16);   // tensor_dim0 lo16
  g1[2] = (int)((((unsigned)chunkElems >> 16) & 0xFFFFu)   // tensor_dim0 hi16
              | (64u << 16));                               // tensor_dim1 lo16
  g1[3] = (int)(((unsigned)chunkElems & 0xFFFFu) << 16);    // tile_dim0
  g1[4] = (int)64u;                                         // tile_dim1=64
  g1[5] = (int)(unsigned)Kpad;                              // dim0_stride lo32
  g1[6] = 0;
  g1[7] = 0;
  i32x4 z4 = {0, 0, 0, 0};
#if __has_include(<hip/amd_detail/amd_gfx1250_TDM.h>)
  i32x8 z8 = {0, 0, 0, 0, 0, 0, 0, 0};
  __builtin_amdgcn_tensor_load_to_lds(g0, g1, z4, z4, z8, 0);
#else
  __builtin_amdgcn_tensor_load_to_lds(g0, g1, z4, z4, 0);
#endif
}

// ------------------------------ tiled GEMM ----------------------------------
// Workgroup: 8 waves, 256 rows x 64 cols. Wave w: strips bm*256+w*16 and
// bm*256+w*16+128; 4 N-tiles each -> 8 accumulators.
// EPI: 0 = bias(+relu) store; 1 = pred min-over-4 rows into x_all;
//      2 = bitmap (+bias)*has_cond into x_all.  K % CHUNK == 0.
template <int CHUNK, int EPI>
__global__ void k_gemm_t(const float* __restrict__ A, int lda, int mstart, int msize,
                         const __bf16* __restrict__ Bt, int Kpad, int K, int Kreal,
                         const float* __restrict__ bias, float* __restrict__ C,
                         int ldc, int coloff, int Nblk, int relu,
                         const int* __restrict__ hc) {
  constexpr int CS = CHUNK * 2 + 16;                 // LDS bytes per column
  __shared__ __align__(16) unsigned char lbuf[2][64 * CS];

  const int bid = blockIdx.x;
  const int bn = bid % Nblk, bm = bid / Nblk;
  const int w = threadIdx.x >> 5, lane = threadIdx.x & 31;
  const int lo = lane & 15, hi = lane >> 4;
  const int rb0 = bm * 256 + w * 16;                 // strip 0
  const int rb1 = rb0 + 128;                         // strip 1
  const int n0 = bn * 64;

  int ar0 = rb0 + lo, ar1 = rb1 + lo;
  if (msize > 0) {
    ar0 = (ar0 / msize) * 15 + mstart + (ar0 % msize);
    ar1 = (ar1 / msize) * 15 + mstart + (ar1 % msize);
  }
  const float* arow0 = A + (size_t)ar0 * lda;
  const float* arow1 = A + (size_t)ar1 * lda;

  const int nch = K / CHUNK;
  constexpr unsigned intv = (CHUNK == 128) ? 5u : 3u;  // 256B / 64B intervals
  const unsigned lds0 = (unsigned)(uintptr_t)(&lbuf[0][0]);
  const unsigned lds1 = (unsigned)(uintptr_t)(&lbuf[1][0]);

  if (threadIdx.x < 32)                               // wave 0 drives the TDM
    tdm_load_panel(lds0, Bt + (size_t)n0 * Kpad, Kpad, CHUNK, intv);

  v8f z = {};
  v8f acc[2][4];
#pragma unroll
  for (int t = 0; t < 4; ++t) { acc[0][t] = z; acc[1][t] = z; }

  for (int c = 0; c < nch; ++c) {
    const int kb = c * CHUNK;
    if (threadIdx.x < 32) {
      if (c + 1 < nch) {
        tdm_load_panel((c & 1) ? lds0 : lds1,
                       Bt + (size_t)n0 * Kpad + (kb + CHUNK), Kpad, CHUNK, intv);
        __builtin_amdgcn_s_wait_tensorcnt(1);         // chunk c complete
      } else {
        __builtin_amdgcn_s_wait_tensorcnt(0);
      }
    }
    __syncthreads();                                  // LDS chunk c visible
    const unsigned char* lb = (c & 1) ? lbuf[1] : lbuf[0];
#pragma unroll
    for (int kk = 0; kk < CHUNK; kk += 32) {
      const int kg = kb + kk;
      v16bf a0, a1;
      if (EPI == 2 && kg + 32 > Kreal) {              // tail guard (bitmap K=1000)
        a0 = load_a_frag_guard(arow0, kg, hi, Kreal);
        a1 = load_a_frag_guard(arow1, kg, hi, Kreal);
      } else {
        __builtin_prefetch(arow0 + kg + 32, 0, 1);
        a0 = load_a_frag(arow0, kg, hi);
        a1 = load_a_frag(arow1, kg, hi);
      }
      uint4 bu[4][2];                                 // hoist all B frags: one
#pragma unroll                                        // ds clause, staged waits
      for (int t = 0; t < 4; ++t) {
        const unsigned char* bp = lb + (t * 16 + lo) * CS + kk * 2 + 16 * hi;
        bu[t][0] = *(const uint4*)bp;
        bu[t][1] = *(const uint4*)(bp + 32);
      }
#pragma unroll
      for (int t = 0; t < 4; ++t) {
        union { uint4 u[2]; v16bf v; } ub;
        ub.u[0] = bu[t][0];
        ub.u[1] = bu[t][1];
        acc[0][t] = __builtin_amdgcn_wmma_f32_16x16x32_bf16(false, a0, false, ub.v,
                                                            (short)0, acc[0][t],
                                                            false, false);
        acc[1][t] = __builtin_amdgcn_wmma_f32_16x16x32_bf16(false, a1, false, ub.v,
                                                            (short)0, acc[1][t],
                                                            false, false);
      }
    }
    __syncthreads();                                  // reads done before reuse
  }

  // ------------------------------- epilogues -------------------------------
#pragma unroll
  for (int s = 0; s < 2; ++s) {
    const int rbase = (s == 0) ? rb0 : rb1;
    if (EPI == 0) {
#pragma unroll
      for (int t = 0; t < 4; ++t) {
        int n = n0 + t * 16 + lo;
        float bv = bias ? bias[n] : 0.0f;
#pragma unroll
        for (int i = 0; i < 8; ++i) {
          int r = rbase + i + 8 * hi;
          float v = acc[s][t][i] + bv;
          if (relu) v = fmaxf(v, 0.0f);
          C[(size_t)r * ldc + coloff + n] = v;
        }
      }
    } else if (EPI == 1) {
      // rows i+8*hi in acc[s][t][i]; min over 4 consecutive pred rows (P dim)
#pragma unroll
      for (int t = 0; t < 4; ++t) {
        int n = n0 + t * 16 + lo;                     // n in [0,128)
        float bv = bias[n];
        float m0 = fminf(fminf(acc[s][t][0], acc[s][t][1]),
                         fminf(acc[s][t][2], acc[s][t][3]));
        float m1 = fminf(fminf(acc[s][t][4], acc[s][t][5]),
                         fminf(acc[s][t][6], acc[s][t][7]));
        int g0 = (rbase >> 2) + 2 * hi;               // pred group index
        int g1 = g0 + 1;                              // (g>>1)=node row, (g&1)=side
        C[(size_t)(g0 >> 1) * 640 + 256 + (g0 & 1) * 128 + n] = m0 + bv;
        C[(size_t)(g1 >> 1) * 640 + 256 + (g1 & 1) * 128 + n] = m1 + bv;
      }
    } else {                                          // EPI == 2 (bitmap)
#pragma unroll
      for (int t = 0; t < 4; ++t) {
        int n = n0 + t * 16 + lo;
        float bv = bias[n];
#pragma unroll
        for (int i = 0; i < 8; ++i) {
          int r = rbase + i + 8 * hi;
          C[(size_t)r * 640 + 512 + n] = (acc[s][t][i] + bv) * (float)hc[r];
        }
      }
    }
  }
}

// ------------------------------ GRU helpers ---------------------------------
__global__ void k_child_avg(const float* __restrict__ hp, float* __restrict__ h0,
                            int S, int total) {
  int i = blockIdx.x * blockDim.x + threadIdx.x;
  if (i >= total) return;
  int c = i & 255;
  int r = i >> 8;
  int b = r / S, s = r - b * S;
  size_t base = ((size_t)(b * 2 * S + 2 * s)) * 256 + c;
  h0[i] = 0.5f * (hp[base] + hp[base + 256]);
}

__global__ void k_gate(const float* __restrict__ Gi, const float* __restrict__ Gh,
                       const float* __restrict__ b_hh, const float* __restrict__ h0,
                       float* __restrict__ hout, int total) {
  int i = blockIdx.x * blockDim.x + threadIdx.x;
  if (i >= total) return;
  int c = i & 255;
  int r = i >> 8;
  size_t gb = (size_t)r * 768 + c;
  float ir = Gi[gb], iz = Gi[gb + 256], in_ = Gi[gb + 512];
  float hr, hz, hn;
  if (Gh) { hr = Gh[gb]; hz = Gh[gb + 256]; hn = Gh[gb + 512]; }
  else    { hr = b_hh[c]; hz = b_hh[c + 256]; hn = b_hh[c + 512]; }
  float h0v = h0 ? h0[i] : 0.0f;
  float rg = sigmoidf_(ir + hr);
  float zg = sigmoidf_(iz + hz);
  float ng = tanhf(in_ + rg * hn);
  hout[i] = (1.0f - zg) * ng + zg * h0v;
}

// ---------------------- head final layer (N=1) + sigmoid --------------------
__global__ void k_head(const float* __restrict__ y2, const float* __restrict__ W3,
                       const float* __restrict__ b3, float* __restrict__ out, int M) {
  int r = blockIdx.x * blockDim.x + threadIdx.x;
  if (r >= M) return;
  float acc = b3[0];
#pragma unroll 8
  for (int j = 0; j < 128; ++j) acc += y2[(size_t)r * 128 + j] * W3[j];
  out[r] = sigmoidf_(acc);
}

// ---------------------------------------------------------------------------
#define LAUNCH_GEMM(CH, EP, A, lda, ms, mz, Bt, Kp, K, Kr, bias, C, ldc, co, M, N, relu, hc) \
  k_gemm_t<CH, EP><<<dim3(((M) / 256) * ((N) / 64)), dim3(256), 0, stream>>>(               \
      A, lda, ms, mz, Bt, Kp, K, Kr, bias, C, ldc, co, (N) / 64, relu, hc)

extern "C" void kernel_launch(void* const* d_in, const int* in_sizes, int n_in,
                              void* d_out, int out_size, void* d_ws, size_t ws_size,
                              hipStream_t stream) {
  const float* op_vecs   = (const float*)d_in[0];
  const float* feat_vecs = (const float*)d_in[1];
  const float* pred_vecs = (const float*)d_in[2];
  const float* bitmap    = (const float*)d_in[3];
  const int*   has_cond  = (const int*)  d_in[4];
  const float* W_op   = (const float*)d_in[5];   const float* b_op   = (const float*)d_in[6];
  const float* W_feat = (const float*)d_in[7];   const float* b_feat = (const float*)d_in[8];
  const float* W_pred = (const float*)d_in[9];   const float* b_pred = (const float*)d_in[10];
  const float* W_bm   = (const float*)d_in[11];  const float* b_bm   = (const float*)d_in[12];
  const float* W_ih   = (const float*)d_in[13];  const float* b_ih   = (const float*)d_in[14];
  const float* W_hh   = (const float*)d_in[15];  const float* b_hh   = (const float*)d_in[16];
  const float* W_c1   = (const float*)d_in[17];  const float* b_c1   = (const float*)d_in[18];
  const float* W_c2   = (const float*)d_in[19];  const float* b_c2   = (const float*)d_in[20];
  const float* W_c3   = (const float*)d_in[21];  const float* b_c3   = (const float*)d_in[22];
  const float* W_k1   = (const float*)d_in[23];  const float* b_k1   = (const float*)d_in[24];
  const float* W_k2   = (const float*)d_in[25];  const float* b_k2   = (const float*)d_in[26];
  const float* W_k3   = (const float*)d_in[27];  const float* b_k3   = (const float*)d_in[28];

  float* outp = (float*)d_out;

  // ---- bump allocator over d_ws ----
  char* ws = (char*)d_ws;
  size_t off = 0;
  auto alloc = [&](size_t bytes) -> void* {
    void* p = ws + off;
    off += (bytes + 255) & ~(size_t)255;
    return p;
  };

  __bf16* wopT   = (__bf16*)alloc((size_t)128 * 32 * 2);
  __bf16* wfeatT = (__bf16*)alloc((size_t)128 * 32 * 2);
  __bf16* wpredT = (__bf16*)alloc((size_t)128 * 128 * 2);
  __bf16* wbmT   = (__bf16*)alloc((size_t)128 * 1024 * 2);
  __bf16* wihT   = (__bf16*)alloc((size_t)768 * 640 * 2);
  __bf16* whhT   = (__bf16*)alloc((size_t)768 * 256 * 2);
  __bf16* wc1T   = (__bf16*)alloc((size_t)128 * 256 * 2);
  __bf16* wc2T   = (__bf16*)alloc((size_t)128 * 128 * 2);
  __bf16* wk1T   = (__bf16*)alloc((size_t)128 * 256 * 2);
  __bf16* wk2T   = (__bf16*)alloc((size_t)128 * 128 * 2);

  float* x_all = (float*)alloc((size_t)MN * 640 * 4);          // 157 MB
  float* Gi    = (float*)alloc((size_t)BSZ * 8 * 768 * 4);     // 101 MB
  float* Gh    = (float*)alloc((size_t)BSZ * 4 * 768 * 4);     // 50 MB
  float* hbuf  = (float*)alloc((size_t)BSZ * 8 * 256 * 4);     // 34 MB
  float* h0b   = (float*)alloc((size_t)BSZ * 4 * 256 * 4);     // 17 MB
  float* y1    = (float*)alloc((size_t)BSZ * 128 * 4);
  float* y2    = (float*)alloc((size_t)BSZ * 128 * 4);
  (void)ws_size; (void)in_sizes; (void)n_in; (void)out_size;

  auto launch_T = [&](const float* src, __bf16* dst, int K, int N, int Kpad) {
    int total = N * Kpad;
    k_wT<<<dim3((total + 255) / 256), dim3(256), 0, stream>>>(src, dst, K, N, Kpad);
  };

  // ---- weight conversion / transposition ----
  launch_T(W_op,   wopT,   32,   128, 32);
  launch_T(W_feat, wfeatT, 32,   128, 32);
  launch_T(W_pred, wpredT, 128,  128, 128);
  launch_T(W_bm,   wbmT,   1000, 128, 1024);
  launch_T(W_ih,   wihT,   640,  768, 640);
  launch_T(W_hh,   whhT,   256,  768, 256);
  launch_T(W_c1,   wc1T,   256,  128, 256);
  launch_T(W_c2,   wc2T,   128,  128, 128);
  launch_T(W_k1,   wk1T,   256,  128, 256);
  launch_T(W_k2,   wk2T,   128,  128, 128);

  // ---- embeddings into x_all (MN x 640) ----
  LAUNCH_GEMM(32, 0, op_vecs,   32, -1, 0, wopT,   32,  32,  32,  b_op,   x_all, 640, 0,   MN, 128, 0, (const int*)nullptr);
  LAUNCH_GEMM(32, 0, feat_vecs, 32, -1, 0, wfeatT, 32,  32,  32,  b_feat, x_all, 640, 128, MN, 128, 0, (const int*)nullptr);
  LAUNCH_GEMM(128, 1, pred_vecs, 128, -1, 0, wpredT, 128, 128, 128, b_pred, x_all, 640, 0, MP, 128, 0, (const int*)nullptr);
  LAUNCH_GEMM(128, 2, bitmap, 1000, -1, 0, wbmT, 1024, 1024, 1000, b_bm, x_all, 640, 0, MN, 128, 0, has_cond);

  // ---- tree GRU, levels d = 3..0 ----
  for (int d = 3; d >= 0; --d) {
    int size  = 1 << d;
    int start = size - 1;
    int M     = BSZ * size;
    // gi = x_level @ W_ih + b_ih  (row map into x_all)
    LAUNCH_GEMM(128, 0, x_all, 640, start, size, wihT, 640, 640, 640, b_ih, Gi, 768, 0, M, 768, 0, (const int*)nullptr);
    const float* h0p = nullptr;
    const float* Ghp = nullptr;
    if (d < 3) {
      int total = M * 256;
      k_child_avg<<<dim3((total + 255) / 256), dim3(256), 0, stream>>>(hbuf, h0b, size, total);
      LAUNCH_GEMM(128, 0, h0b, 256, -1, 0, whhT, 256, 256, 256, b_hh, Gh, 768, 0, M, 768, 0, (const int*)nullptr);
      h0p = h0b; Ghp = Gh;
    }
    int total = M * 256;
    k_gate<<<dim3((total + 255) / 256), dim3(256), 0, stream>>>(Gi, Ghp, b_hh, h0p, hbuf, total);
  }
  // root = hbuf (BSZ x 256)

  // ---- cost head ----
  LAUNCH_GEMM(128, 0, hbuf, 256, -1, 0, wc1T, 256, 256, 256, b_c1, y1, 128, 0, BSZ, 128, 1, (const int*)nullptr);
  LAUNCH_GEMM(128, 0, y1,   128, -1, 0, wc2T, 128, 128, 128, b_c2, y2, 128, 0, BSZ, 128, 1, (const int*)nullptr);
  k_head<<<dim3((BSZ + 255) / 256), dim3(256), 0, stream>>>(y2, W_c3, b_c3, outp, BSZ);

  // ---- card head ----
  LAUNCH_GEMM(128, 0, hbuf, 256, -1, 0, wk1T, 256, 256, 256, b_k1, y1, 128, 0, BSZ, 128, 1, (const int*)nullptr);
  LAUNCH_GEMM(128, 0, y1,   128, -1, 0, wk2T, 128, 128, 128, b_k2, y2, 128, 0, BSZ, 128, 1, (const int*)nullptr);
  k_head<<<dim3((BSZ + 255) / 256), dim3(256), 0, stream>>>(y2, W_k3, b_k3, outp + BSZ, BSZ);
}
